// V_FNO_DDP_38671885533651
// MI455X (gfx1250) — matrix operand order repaired and checked by reading, not compiled, not run on hardware
//
#include <hip/hip_runtime.h>
#include <hip/hip_bf16.h>
#include <math.h>

// ---------------- constants ----------------
constexpr int kB  = 256;
constexpr int kSX = 179;
constexpr int kSY = 15;
constexpr int kW  = 26;
constexpr int kM  = 7;
constexpr int kNP = kSX * kSY;        // 2685 pixels
constexpr int kNKX = 14;              // kept x-modes (0..6, 172..178)
constexpr int kNK  = kNKX * kM;       // 98 kept modes
constexpr int kKF  = kW * kNP;        // 69810 = flat feature length
constexpr int kKP  = 69824;           // padded to multiple of 32
constexpr int kN1  = 2808;            // o1 hidden size
constexpr float kTWOPI = 6.28318530717958647692f;

typedef __bf16 bf16_t;
typedef bf16_t v16bf __attribute__((ext_vector_type(16)));
typedef float  v8f   __attribute__((ext_vector_type(8)));
typedef float  f32x2 __attribute__((ext_vector_type(2)));
typedef unsigned int u32x4 __attribute__((ext_vector_type(4)));

// ---------------- helpers ----------------
__device__ __forceinline__ void reduce2(float& s1, float& s2, float* r1, float* r2) {
  const int tid = threadIdx.x;
  r1[tid] = s1; r2[tid] = s2;
  __syncthreads();
  for (int off = 128; off > 0; off >>= 1) {
    if (tid < off) { r1[tid] += r1[tid + off]; r2[tid] += r2[tid + off]; }
    __syncthreads();
  }
  s1 = r1[0]; s2 = r2[0];
  __syncthreads();
}

__device__ __forceinline__ float gelu_exact(float x) {
  return 0.5f * x * (1.0f + erff(x * 0.70710678118654752440f));
}

// ---------------- small front-end kernels ----------------
__global__ void k_x0(const float* __restrict__ x, float* __restrict__ x0) {
  const int idx = blockIdx.x * 256 + threadIdx.x;
  if (idx >= kB * 128) return;
  const int b = idx >> 7, j = idx & 127;
  x0[idx] = (x[((size_t)b * kSX + j) * (kSY + 1) + kSY] - 400.0f) * 0.01f;
}

__global__ __launch_bounds__(256)
void k_embed(const float* __restrict__ x, const float* __restrict__ pw,
             const float* __restrict__ pb, float* __restrict__ h) {
  __shared__ float spw[kW * 3], spb[kW];
  const int tid = threadIdx.x;
  if (tid < kW * 3) spw[tid] = pw[tid];
  if (tid < kW)     spb[tid] = pb[tid];
  __syncthreads();
  const int idx = blockIdx.x * 256 + tid;
  if (idx >= kB * kNP) return;
  const int b = idx / kNP, p = idx % kNP;
  const int xx = p / kSY, yy = p % kSY;
  const float u  = x[((size_t)b * kSX + xx) * (kSY + 1) + yy];
  const float gx = (float)xx * (1.0f / (float)(kSX - 1));
  const float gy = (float)yy * (1.0f / (float)(kSY - 1));
#pragma unroll
  for (int c = 0; c < kW; ++c)
    h[((size_t)b * kW + c) * kNP + p] =
        spw[c * 3] * u + spw[c * 3 + 1] * gx + spw[c * 3 + 2] * gy + spb[c];
}

// ---------------- InstanceNorm + forward truncated DFT (per (b,c)) ----------------
__global__ __launch_bounds__(256)
void k_fwd(const float* __restrict__ src, float* __restrict__ modes) {
  __shared__ float sh[kNP];
  __shared__ float tyr[kSX * kM], tyi[kSX * kM];
  __shared__ float cX[kNKX * kSX], sX[kNKX * kSX];
  __shared__ float cY[kM * kSY],   sY[kM * kSY];
  __shared__ float r1[256], r2[256];
  const int tid = threadIdx.x;
  const int bc  = blockIdx.x;
  const float* in = src + (size_t)bc * kNP;

  float s = 0.f, s2 = 0.f;
  for (int i = tid; i < kNP; i += 256) {
    float v = in[i];
    sh[i] = v; s += v; s2 += v * v;
  }
  for (int i = tid; i < kNKX * kSX; i += 256) {
    int kxi = i / kSX, xx = i % kSX;
    int kx = kxi < 7 ? kxi : kxi + 165;
    float a = -kTWOPI * (float)((kx * xx) % kSX) / (float)kSX;
    float sn, cc; sincosf(a, &sn, &cc);
    cX[i] = cc; sX[i] = sn;
  }
  for (int i = tid; i < kM * kSY; i += 256) {
    int ky = i / kSY, yy = i % kSY;
    float a = -kTWOPI * (float)((ky * yy) % kSY) / (float)kSY;
    float sn, cc; sincosf(a, &sn, &cc);
    cY[i] = cc; sY[i] = sn;
  }
  reduce2(s, s2, r1, r2);
  const float mean = s * (1.0f / (float)kNP);
  const float var  = s2 * (1.0f / (float)kNP) - mean * mean;
  const float rstd = rsqrtf(var + 1e-5f);
  for (int i = tid; i < kNP; i += 256) sh[i] = (sh[i] - mean) * rstd;
  __syncthreads();
  // DFT over y (15 -> 7 modes) per row
  for (int o = tid; o < kSX * kM; o += 256) {
    int xx = o / kM, ky = o % kM;
    const float* row = sh + xx * kSY;
    float re = 0.f, im = 0.f;
#pragma unroll
    for (int y = 0; y < kSY; ++y) {
      float v = row[y];
      re += v * cY[ky * kSY + y];
      im += v * sY[ky * kSY + y];
    }
    tyr[o] = re; tyi[o] = im;
  }
  __syncthreads();
  // DFT over x (179 -> 14 kept modes)
  for (int o = tid; o < kNK; o += 256) {
    int kxi = o / kM, ky = o % kM;
    float re = 0.f, im = 0.f;
    for (int xx = 0; xx < kSX; ++xx) {
      float c = cX[kxi * kSX + xx], sn = sX[kxi * kSX + xx];
      float tr = tyr[xx * kM + ky], ti = tyi[xx * kM + ky];
      re += tr * c - ti * sn;
      im += tr * sn + ti * c;
    }
    modes[((size_t)bc * kNK + o) * 2]     = re;
    modes[((size_t)bc * kNK + o) * 2 + 1] = im;
  }
}

// ---------------- complex channel mix over kept modes ----------------
__global__ void k_mix(const float* __restrict__ mA, const float* __restrict__ w1,
                      const float* __restrict__ w2, float* __restrict__ mB) {
  const int idx = blockIdx.x * 256 + threadIdx.x;
  if (idx >= kB * kW * kNK) return;
  const int m  = idx % kNK;
  const int co = (idx / kNK) % kW;
  const int b  = idx / (kNK * kW);
  const int kxi = m / kM, ky = m % kM;
  const float* wsel = (kxi < 7) ? w1 : w2;
  const int kxw = (kxi < 7) ? kxi : kxi - 7;
  float re = 0.f, im = 0.f;
#pragma unroll
  for (int ci = 0; ci < kW; ++ci) {
    const float* a  = mA + (((size_t)b * kW + ci) * kNK + m) * 2;
    const float* wv = wsel + (((size_t)(ci * kW + co) * kM + kxw) * kM + ky) * 2;
    float ar = a[0], ai = a[1], wr = wv[0], wi = wv[1];
    re += ar * wr - ai * wi;
    im += ar * wi + ai * wr;
  }
  float* o = mB + (((size_t)b * kW + co) * kNK + m) * 2;
  o[0] = re; o[1] = im;
}

// ---------------- inverse truncated DFT + InstanceNorm (per (b,c)) ----------------
__global__ __launch_bounds__(256)
void k_inv(const float* __restrict__ modes, float* __restrict__ dst) {
  __shared__ float mr[kNK], mi[kNK];
  __shared__ float tre[kSX * kM], tim[kSX * kM];
  __shared__ float sh[kNP];
  __shared__ float cX[kNKX * kSX], sX[kNKX * kSX];
  __shared__ float cY[kM * kSY],   sY[kM * kSY];
  __shared__ float r1[256], r2[256];
  const int tid = threadIdx.x;
  const int bc  = blockIdx.x;
  const float* src = modes + (size_t)bc * kNK * 2;
  for (int i = tid; i < kNK; i += 256) { mr[i] = src[2 * i]; mi[i] = src[2 * i + 1]; }
  for (int i = tid; i < kNKX * kSX; i += 256) {
    int kxi = i / kSX, xx = i % kSX;
    int kx = kxi < 7 ? kxi : kxi + 165;
    float a = kTWOPI * (float)((kx * xx) % kSX) / (float)kSX;
    float sn, cc; sincosf(a, &sn, &cc);
    cX[i] = cc; sX[i] = sn;
  }
  for (int i = tid; i < kM * kSY; i += 256) {
    int ky = i / kSY, yy = i % kSY;
    float a = kTWOPI * (float)((ky * yy) % kSY) / (float)kSY;
    float sn, cc; sincosf(a, &sn, &cc);
    cY[i] = cc; sY[i] = sn;
  }
  __syncthreads();
  // inverse FFT over x (14 kept modes -> 179 positions)
  for (int o = tid; o < kSX * kM; o += 256) {
    int xx = o / kM, ky = o % kM;
    float re = 0.f, im = 0.f;
#pragma unroll
    for (int kxi = 0; kxi < kNKX; ++kxi) {
      float c = cX[kxi * kSX + xx], sn = sX[kxi * kSX + xx];
      float ar = mr[kxi * kM + ky], ai = mi[kxi * kM + ky];
      re += ar * c - ai * sn;
      im += ar * sn + ai * c;
    }
    tre[o] = re * (1.0f / (float)kSX);
    tim[o] = im * (1.0f / (float)kSX);
  }
  __syncthreads();
  // irfft over y (Hermitian extension of 7 modes -> 15)
  float s = 0.f, s2 = 0.f;
  for (int p = tid; p < kNP; p += 256) {
    int xx = p / kSY, yy = p % kSY;
    float acc = tre[xx * kM];
#pragma unroll
    for (int ky = 1; ky < kM; ++ky)
      acc += 2.0f * (tre[xx * kM + ky] * cY[ky * kSY + yy] -
                     tim[xx * kM + ky] * sY[ky * kSY + yy]);
    float v = acc * (1.0f / (float)kSY);
    sh[p] = v; s += v; s2 += v * v;
  }
  reduce2(s, s2, r1, r2);
  const float mean = s * (1.0f / (float)kNP);
  const float var  = s2 * (1.0f / (float)kNP) - mean * mean;
  const float rstd = rsqrtf(var + 1e-5f);
  float* out = dst + (size_t)bc * kNP;
  for (int p = tid; p < kNP; p += 256) out[p] = (sh[p] - mean) * rstd;
}

// ---------------- fused MLP(1x1) + skip 1x1 + GELU (in-place on h) ----------------
__global__ __launch_bounds__(256)
void k_mlp(const float* __restrict__ t, float* __restrict__ h,
           const float* __restrict__ w1, const float* __restrict__ b1,
           const float* __restrict__ w2, const float* __restrict__ b2,
           const float* __restrict__ w0, const float* __restrict__ b0) {
  __shared__ float sw1[kW * kW], sw2[kW * kW], sw0[kW * kW];
  __shared__ float sb1[kW], sb2[kW], sb0[kW];
  const int tid = threadIdx.x;
  for (int i = tid; i < kW * kW; i += 256) { sw1[i] = w1[i]; sw2[i] = w2[i]; sw0[i] = w0[i]; }
  if (tid < kW) { sb1[tid] = b1[tid]; sb2[tid] = b2[tid]; sb0[tid] = b0[tid]; }
  __syncthreads();
  const int idx = blockIdx.x * 256 + tid;
  if (idx >= kB * kNP) return;
  const int b = idx / kNP, p = idx % kNP;
  float tv[kW], hv[kW], m1[kW];
#pragma unroll
  for (int c = 0; c < kW; ++c) {
    tv[c] = t[((size_t)b * kW + c) * kNP + p];
    hv[c] = h[((size_t)b * kW + c) * kNP + p];
  }
#pragma unroll
  for (int o = 0; o < kW; ++o) {
    float a = sb1[o];
#pragma unroll
    for (int i = 0; i < kW; ++i) a += sw1[o * kW + i] * tv[i];
    m1[o] = gelu_exact(a);
  }
#pragma unroll
  for (int o = 0; o < kW; ++o) {
    float a  = sb2[o];
    float sk = sb0[o];
#pragma unroll
    for (int i = 0; i < kW; ++i) { a += sw2[o * kW + i] * m1[i]; sk += sw0[o * kW + i] * hv[i]; }
    h[((size_t)b * kW + o) * kNP + p] = gelu_exact(a + sk);
  }
}

// ---------------- spatial attention: conv3x3 -> relu -> GAP -> fc -> sigmoid ----------------
__global__ __launch_bounds__(256)
void k_att(const float* __restrict__ h, const float* __restrict__ cw,
           const float* __restrict__ cb, const float* __restrict__ fw,
           const float* __restrict__ fb, float* __restrict__ att) {
  __shared__ float sw[kW * kW * 9];
  __shared__ float scb[kW];
  __shared__ float red[256];
  __shared__ float chans[kW];
  const int tid = threadIdx.x;
  const int b = blockIdx.x;
  for (int i = tid; i < kW * kW * 9; i += 256) sw[i] = cw[i];
  if (tid < kW) scb[tid] = cb[tid];
  __syncthreads();
  float sum[kW];
#pragma unroll
  for (int c = 0; c < kW; ++c) sum[c] = 0.f;
  for (int p = tid; p < kNP; p += 256) {
    const int xx = p / kSY, yy = p % kSY;
    float acc[kW];
#pragma unroll
    for (int c = 0; c < kW; ++c) acc[c] = scb[c];
    for (int ci = 0; ci < kW; ++ci) {
      const float* base = h + ((size_t)b * kW + ci) * kNP;
      for (int dx = -1; dx <= 1; ++dx) {
        int x2 = xx + dx; if (x2 < 0 || x2 >= kSX) continue;
        for (int dy = -1; dy <= 1; ++dy) {
          int y2 = yy + dy; if (y2 < 0 || y2 >= kSY) continue;
          float v = base[x2 * kSY + y2];
          int wi = ci * 9 + (dx + 1) * 3 + (dy + 1);
#pragma unroll
          for (int co = 0; co < kW; ++co) acc[co] += sw[co * (kW * 9) + wi] * v;
        }
      }
    }
#pragma unroll
    for (int c = 0; c < kW; ++c) sum[c] += acc[c] > 0.f ? acc[c] : 0.f;
  }
  for (int c = 0; c < kW; ++c) {
    red[tid] = sum[c];
    __syncthreads();
    for (int off = 128; off > 0; off >>= 1) {
      if (tid < off) red[tid] += red[tid + off];
      __syncthreads();
    }
    if (tid == 0) chans[c] = red[0];
    __syncthreads();
  }
  if (tid == 0) {
    float fc = fb[0];
#pragma unroll
    for (int c = 0; c < kW; ++c) fc += (chans[c] * (1.0f / (float)kNP)) * fw[c];
    att[b] = 1.0f / (1.0f + expf(-fc));
  }
}

// ---------------- scale by attention + cast to zero-padded bf16 A-matrix ----------------
__global__ void k_cast(const float* __restrict__ h, const float* __restrict__ att,
                       bf16_t* __restrict__ flat) {
  const size_t idx = (size_t)blockIdx.x * 256 + threadIdx.x;
  if (idx >= (size_t)kB * kKP) return;
  const int b = (int)(idx / kKP);
  const int k = (int)(idx % kKP);
  float v = (k < kKF) ? h[(size_t)b * kKF + k] * att[b] : 0.0f;
  flat[idx] = (bf16_t)v;
}

// ---------------- big GEMM: (256 x 69810) x (69810 x 2808), bf16 WMMA, f32 acc ----------------
// stream f32 weights from HBM once, convert to bf16 in registers, fuse bias+leaky_relu.
__global__ __launch_bounds__(128)
void k_gemm1(const bf16_t* __restrict__ A, const float* __restrict__ W1,
             const float* __restrict__ b1, float* __restrict__ out) {
  const int tid  = threadIdx.x;
  const int gw   = blockIdx.x * 4 + (tid >> 5);   // global wave id = one 16x16 tile
  const int rt   = gw & 15;                        // row tile (256/16)
  const int ct   = gw >> 4;                        // col tile (2816/16)
  const int lane = tid & 31;
  const int half = lane >> 4;
  const int l15  = lane & 15;

  const int row = rt * 16 + l15;
  const bf16_t* arow = A + (size_t)row * kKP;
  const int n  = ct * 16 + l15;
  const int nc = n < kN1 ? n : (kN1 - 1);          // clamp: garbage only feeds masked cols
  const float* wb = W1 + (size_t)nc * kKF;

  v8f acc = {0.f, 0.f, 0.f, 0.f, 0.f, 0.f, 0.f, 0.f};

  for (int kt = 0; kt < 2181; ++kt) {
    const int k0 = kt * 32;
    if ((kt & 15) == 0) __builtin_prefetch(wb + k0 + 2048, 0, 0);
    union { u32x4 u[2]; v16bf v; } a;
    const bf16_t* ap = arow + k0 + half * 16;      // contiguous k = 16*half + s (matches B)
    a.u[0] = *(const u32x4*)(ap);
    a.u[1] = *(const u32x4*)(ap + 8);
    const float* wp = wb + k0 + half * 16;         // contiguous k = 16*half + s (matches A)
    v16bf bv;
#pragma unroll
    for (int j = 0; j < 8; ++j) {
      f32x2 q = *(const f32x2*)(wp + 2 * j);
      bv[2 * j]     = (bf16_t)q.x;
      bv[2 * j + 1] = (bf16_t)q.y;
    }
    acc = __builtin_amdgcn_wmma_f32_16x16x32_bf16(false, a.v, false, bv,
                                                  (short)0, acc, false, false);
  }
  { // K tail: k0 = 69792 .. 69823, guard k < 69810 (A pad is zeroed)
    const int k0 = 2181 * 32;
    union { u32x4 u[2]; v16bf v; } a;
    const bf16_t* ap = arow + k0 + half * 16;
    a.u[0] = *(const u32x4*)(ap);
    a.u[1] = *(const u32x4*)(ap + 8);
    v16bf bv;
#pragma unroll
    for (int i = 0; i < 16; ++i) {
      int k = k0 + half * 16 + i;
      bv[i] = (bf16_t)(k < kKF ? wb[k] : 0.0f);
    }
    acc = __builtin_amdgcn_wmma_f32_16x16x32_bf16(false, a.v, false, bv,
                                                  (short)0, acc, false, false);
  }
  if (n < kN1) {
    const float bias = b1[n];
#pragma unroll
    for (int v = 0; v < 8; ++v) {
      const int r = rt * 16 + v + half * 8;        // D layout: VGPR v -> M=v (+8 for hi lanes)
      float val = acc[v] + bias;
      out[(size_t)r * kN1 + n] = val > 0.f ? val : 0.01f * val;   // leaky_relu
    }
  }
}

// ---------------- small GEMM: (256 x 2808) x (2808 x 128) ----------------
__global__ __launch_bounds__(128)
void k_gemm2(const float* __restrict__ out1, const float* __restrict__ w,
             const float* __restrict__ bias, float* __restrict__ x1) {
  __shared__ float srow[kN1];
  const int b = blockIdx.x, tid = threadIdx.x;
  for (int i = tid; i < kN1; i += 128) srow[i] = out1[(size_t)b * kN1 + i];
  __syncthreads();
  float acc = bias[tid];
  const float* wr = w + (size_t)tid * kN1;
  for (int n = 0; n < kN1; ++n) acc += srow[n] * wr[n];
  x1[b * 128 + tid] = acc;
}

// ---------------- interleave x1/x0 and final regression ----------------
__global__ void k_final(const float* __restrict__ x1, const float* __restrict__ x0,
                        const float* __restrict__ rw, const float* __restrict__ rb,
                        float* __restrict__ out) {
  const int b = blockIdx.x * 256 + threadIdx.x;
  if (b >= kB) return;
  float acc = rb[0];
#pragma unroll
  for (int j = 0; j < 128; ++j)
    acc += rw[2 * j] * x1[b * 128 + j] + rw[2 * j + 1] * x0[b * 128 + j];
  out[b] = acc;
}

// ---------------- host launch ----------------
extern "C" void kernel_launch(void* const* d_in, const int* in_sizes, int n_in,
                              void* d_out, int out_size, void* d_ws, size_t ws_size,
                              hipStream_t stream) {
  (void)in_sizes; (void)n_in; (void)out_size; (void)ws_size;
  const float* x     = (const float*)d_in[0];
  const float* p_w   = (const float*)d_in[1];
  const float* p_b   = (const float*)d_in[2];
  const float* sc0w1 = (const float*)d_in[3];
  const float* sc0w2 = (const float*)d_in[4];
  const float* sc1w1 = (const float*)d_in[5];
  const float* sc1w2 = (const float*)d_in[6];
  const float* m0w1  = (const float*)d_in[7];
  const float* m0b1  = (const float*)d_in[8];
  const float* m0w2  = (const float*)d_in[9];
  const float* m0b2  = (const float*)d_in[10];
  const float* m1w1  = (const float*)d_in[11];
  const float* m1b1  = (const float*)d_in[12];
  const float* m1w2  = (const float*)d_in[13];
  const float* m1b2  = (const float*)d_in[14];
  const float* w0w   = (const float*)d_in[15];
  const float* w0b   = (const float*)d_in[16];
  const float* w1w   = (const float*)d_in[17];
  const float* w1b   = (const float*)d_in[18];
  const float* sacw  = (const float*)d_in[19];
  const float* sacb  = (const float*)d_in[20];
  const float* safw  = (const float*)d_in[21];
  const float* safb  = (const float*)d_in[22];
  const float* o1w1  = (const float*)d_in[23];
  const float* o1b1  = (const float*)d_in[24];
  const float* o1w2  = (const float*)d_in[25];
  const float* o1b2  = (const float*)d_in[26];
  const float* reg2w = (const float*)d_in[27];
  const float* reg2b = (const float*)d_in[28];

  char* ws = (char*)d_ws;
  size_t off = 0;
  auto take = [&](size_t bytes) -> char* {
    char* p = ws + off;
    off = (off + bytes + 255) & ~(size_t)255;
    return p;
  };
  float*  bufH   = (float*)take((size_t)kB * kW * kNP * 4);       // 71.5 MB
  float*  bufT   = (float*)take((size_t)kB * kW * kNP * 4);       // 71.5 MB
  float*  modesA = (float*)take((size_t)kB * kW * kNK * 2 * 4);   // 5.2 MB
  float*  modesB = (float*)take((size_t)kB * kW * kNK * 2 * 4);   // 5.2 MB
  bf16_t* flat   = (bf16_t*)take((size_t)kB * kKP * 2);           // 35.7 MB
  float*  out1   = (float*)take((size_t)kB * kN1 * 4);            // 2.9 MB
  float*  x0b    = (float*)take((size_t)kB * 128 * 4);
  float*  x1b    = (float*)take((size_t)kB * 128 * 4);
  float*  attb   = (float*)take((size_t)kB * 4);

  const int nBlocksNP = (kB * kNP + 255) / 256;      // 2685
  const int nBlocksMx = (kB * kW * kNK + 255) / 256; // 2548

  k_x0<<<(kB * 128 + 255) / 256, 256, 0, stream>>>(x, x0b);
  k_embed<<<nBlocksNP, 256, 0, stream>>>(x, p_w, p_b, bufH);

  // Fourier block 0
  k_fwd<<<kB * kW, 256, 0, stream>>>(bufH, modesA);
  k_mix<<<nBlocksMx, 256, 0, stream>>>(modesA, sc0w1, sc0w2, modesB);
  k_inv<<<kB * kW, 256, 0, stream>>>(modesB, bufT);
  k_mlp<<<nBlocksNP, 256, 0, stream>>>(bufT, bufH, m0w1, m0b1, m0w2, m0b2, w0w, w0b);

  // Fourier block 1
  k_fwd<<<kB * kW, 256, 0, stream>>>(bufH, modesA);
  k_mix<<<nBlocksMx, 256, 0, stream>>>(modesA, sc1w1, sc1w2, modesB);
  k_inv<<<kB * kW, 256, 0, stream>>>(modesB, bufT);
  k_mlp<<<nBlocksNP, 256, 0, stream>>>(bufT, bufH, m1w1, m1b1, m1w2, m1b2, w1w, w1b);

  // attention, scale+cast, big WMMA GEMM, tail GEMMs
  k_att<<<kB, 256, 0, stream>>>(bufH, sacw, sacb, safw, safb, attb);
  k_cast<<<(int)(((size_t)kB * kKP + 255) / 256), 256, 0, stream>>>(bufH, attb, flat);
  k_gemm1<<<704, 128, 0, stream>>>(flat, o1w1, o1b1, out1);   // 704*4 waves = 16*176 tiles
  k_gemm2<<<kB, 128, 0, stream>>>(out1, o1w2, o1b2, x1b);
  k_final<<<1, 256, 0, stream>>>(x1b, x0b, reg2w, reg2b, (float*)d_out);
}
